// WindowSelfAttention_30279519436911
// MI455X (gfx1250) — matrix operand order repaired and checked by reading, not compiled, not run on hardware
//
#include <hip/hip_runtime.h>

// ---------------------------------------------------------------------------
// Problem constants (from reference): B=2, H=W=224, DIM=768, NH=12, HD=64,
// window 7x7 (L=49), shift=3, tau=0.01, rope base=100.
// ---------------------------------------------------------------------------
#define DIMC   768
#define NHEAD  12
#define HDIM   64
#define WIN    7
#define LWIN   49
#define HWDIM  224
#define NWSIDE 32            // 224/7 windows per side
#define NBATCH 2
#define NTOK   (NBATCH * HWDIM * HWDIM)   // 100352
#define QKVN   (3 * DIMC)                 // 2304
#define SHFT   3

typedef __attribute__((ext_vector_type(8)))  float  v8f;
typedef __attribute__((ext_vector_type(16))) __bf16 v16bf;
typedef __attribute__((ext_vector_type(8)))  __bf16 v8bf;

// ---------------------------------------------------------------------------
// bf16 helpers (RNE convert)
// ---------------------------------------------------------------------------
__device__ __forceinline__ unsigned short f2bf(float f) {
  union { float f; unsigned u; } v; v.f = f;
  unsigned u = v.u;
  u += 0x7fffu + ((u >> 16) & 1u);
  return (unsigned short)(u >> 16);
}
__device__ __forceinline__ float bf2f(unsigned short h) {
  union { unsigned u; float f; } v; v.u = ((unsigned)h) << 16;
  return v.f;
}
__device__ __forceinline__ unsigned pack2(unsigned short a, unsigned short b) {
  return (unsigned)a | ((unsigned)b << 16);
}

// ---------------------------------------------------------------------------
// CDNA5 async global->LDS copy (ASYNCcnt-tracked; ISA 10. "Async Memory").
// Each active lane copies 16 bytes from its global address to its LDS offset.
// ---------------------------------------------------------------------------
__device__ __forceinline__ void async_b128(unsigned lds_byte_off,
                                           const void* gaddr) {
  asm volatile("global_load_async_to_lds_b128 %0, %1, off"
               :: "v"(lds_byte_off), "v"((unsigned long long)(uintptr_t)gaddr)
               : "memory");
}
__device__ __forceinline__ void wait_async4() {
  asm volatile("s_wait_asynccnt 0x4" ::: "memory");
}
__device__ __forceinline__ void wait_async0() {
  asm volatile("s_wait_asynccnt 0x0" ::: "memory");
}

// ---------------------------------------------------------------------------
// WMMA fragment loads from LDS (bf16, 16x16x32 shapes, wave32).
// A (16x32): lane l -> row (l&15); elems 0-7 = K kb..kb+7, elems 8-15 =
//            K kb+16..kb+23, with kb=(l>>4)*8.               [ISA 7.12.2]
// B (32x16): lane l -> col (l&15); same K order as A (kb=(l>>4)*8).
// Each load is two 16-byte LDS reads (ds_load_b128).
// ---------------------------------------------------------------------------
__device__ __forceinline__ v16bf ld_frag_a(const unsigned short* lds, int row,
                                           int stride, int kofs) {
  const v8bf lo = *(const v8bf*)(lds + row * stride + kofs);
  const v8bf hi = *(const v8bf*)(lds + row * stride + kofs + 16);
  return __builtin_shufflevector(lo, hi, 0, 1, 2, 3, 4, 5, 6, 7,
                                 8, 9, 10, 11, 12, 13, 14, 15);
}
__device__ __forceinline__ v16bf ld_frag_b(const unsigned short* lds, int col,
                                           int stride, int kofs) {
  const v8bf lo = *(const v8bf*)(lds + col * stride + kofs);
  const v8bf hi = *(const v8bf*)(lds + col * stride + kofs + 16);
  return __builtin_shufflevector(lo, hi, 0, 1, 2, 3, 4, 5, 6, 7,
                                 8, 9, 10, 11, 12, 13, 14, 15);
}
__device__ __forceinline__ v8f wmma_bf16(v16bf a, v16bf b, v8f c) {
  return __builtin_amdgcn_wmma_f32_16x16x32_bf16(false, a, false, b,
                                                 (short)0, c, false, false);
}

// ---------------------------------------------------------------------------
// fp32 -> bf16 bulk conversion (one-time pass so GEMM operands are pure
// byte-copies and A-traffic is halved).
// ---------------------------------------------------------------------------
__global__ __launch_bounds__(256) void cvt_bf16(const float* __restrict__ src,
                                                unsigned short* __restrict__ dst,
                                                long n) {
  const long i = ((long)blockIdx.x * 256 + threadIdx.x) * 8;
  if (i + 8 <= n) {
    const float4* p = (const float4*)(src + i);
    const float4 a = p[0], b = p[1];
    *(uint4*)(dst + i) =
        make_uint4(pack2(f2bf(a.x), f2bf(a.y)), pack2(f2bf(a.z), f2bf(a.w)),
                   pack2(f2bf(b.x), f2bf(b.y)), pack2(f2bf(b.z), f2bf(b.w)));
  }
}

// ---------------------------------------------------------------------------
// Tiled GEMM:  Out = A[M x K](bf16) * B[N x K](bf16)^T
// Block tile 128x128, K-tile 32, double-buffered async global->LDS staging.
// 256 threads = 8 waves; each wave computes a 32x64 sub-tile = 2x4 WMMA accs.
// ---------------------------------------------------------------------------
template <int NSTRIDE, bool OUT_IS_BF16>
__global__ __launch_bounds__(256) void gemm_bt(
    const unsigned short* __restrict__ A, const unsigned short* __restrict__ Bw,
    void* __restrict__ Outv, int K) {
  __shared__ __align__(16) unsigned short sA[2][128 * 40];
  __shared__ __align__(16) unsigned short sB[2][128 * 40];

  const int t    = threadIdx.x;
  const int lane = t & 31;
  const int wv   = t >> 5;
  const int wr   = wv >> 1;  // wave row group (0..3) -> rows wr*32..
  const int wc   = wv & 1;   // wave col group (0..1) -> cols wc*64..

  const int srow = t >> 1;        // staging row 0..127
  const int skb  = (t & 1) * 16;  // staging k offset 0 / 16

  const long gA = ((long)blockIdx.x * 128 + srow) * K + skb;
  const long gB = ((long)blockIdx.y * 128 + srow) * K + skb;

  const unsigned aoff[2] = {
      (unsigned)(uintptr_t)(&sA[0][srow * 40 + skb]),
      (unsigned)(uintptr_t)(&sA[1][srow * 40 + skb])};
  const unsigned boff[2] = {
      (unsigned)(uintptr_t)(&sB[0][srow * 40 + skb]),
      (unsigned)(uintptr_t)(&sB[1][srow * 40 + skb])};

  const int l15  = lane & 15;
  const int kb8  = (lane >> 4) * 8;
  const int kb16 = (lane >> 4) * 16;

  v8f acc[2][4];
#pragma unroll
  for (int i = 0; i < 2; ++i)
#pragma unroll
    for (int j = 0; j < 4; ++j) acc[i][j] = (v8f)(0.0f);

  // prologue: stage K-tile 0 into buffer 0 (4 async b128 per thread)
  async_b128(aoff[0], A + gA);
  async_b128(aoff[0] + 16, A + gA + 8);
  async_b128(boff[0], Bw + gB);
  async_b128(boff[0] + 16, Bw + gB + 8);

  const int NT = K / 32;
  for (int kt = 0; kt < NT; ++kt) {
    const int cur = kt & 1;
    const int k0n = (kt + 1) * 32;
    if (k0n < K) {  // prefetch next K-tile into the other buffer
      const int nxt = cur ^ 1;
      async_b128(aoff[nxt], A + gA + k0n);
      async_b128(aoff[nxt] + 16, A + gA + k0n + 8);
      async_b128(boff[nxt], Bw + gB + k0n);
      async_b128(boff[nxt] + 16, Bw + gB + k0n + 8);
      wait_async4();  // tile kt complete; only the 4 new loads outstanding
      __builtin_prefetch(A + gA + k0n + 32, 0, 1);   // L2 warm (k+2)
      __builtin_prefetch(Bw + gB + k0n + 32, 0, 1);
    } else {
      wait_async0();
    }
    __syncthreads();

    v16bf af[2];
#pragma unroll
    for (int a2 = 0; a2 < 2; ++a2)
      af[a2] = ld_frag_a(sA[cur], wr * 32 + a2 * 16 + l15, 40, kb8);
    v16bf bfr[4];
#pragma unroll
    for (int b2 = 0; b2 < 4; ++b2)
      bfr[b2] = ld_frag_b(sB[cur], wc * 64 + b2 * 16 + l15, 40, kb8);

#pragma unroll
    for (int a2 = 0; a2 < 2; ++a2)
#pragma unroll
      for (int b2 = 0; b2 < 4; ++b2)
        acc[a2][b2] = wmma_bf16(af[a2], bfr[b2], acc[a2][b2]);

    __syncthreads();  // everyone done with buf[cur] before it is re-filled
  }

  // ---- epilogue: C layout = VGPR j -> row j (lanes 0-15) / j+8 (lanes 16-31)
  const long rbase = (long)blockIdx.x * 128 + wr * 32 + ((lane >> 4) << 3);
  const int  cbase = blockIdx.y * 128 + wc * 64 + l15;
#pragma unroll
  for (int a2 = 0; a2 < 2; ++a2) {
#pragma unroll
    for (int b2 = 0; b2 < 4; ++b2) {
#pragma unroll
      for (int j = 0; j < 8; ++j) {
        const long R = rbase + a2 * 16 + j;
        const int  C = cbase + b2 * 16;
        const float val = acc[a2][b2][j];
        if constexpr (OUT_IS_BF16)
          ((unsigned short*)Outv)[R * NSTRIDE + C] = f2bf(val);
        else
          ((float*)Outv)[R * NSTRIDE + C] = val;
      }
    }
  }
}

// ---------------------------------------------------------------------------
// Attention kernel: one block (128 threads = 4 waves) per (window, head).
// Gathers q/k/v (with shift roll), applies RoPE + L2 norm, runs
// softmax(Q K^T / tau) V with WMMA on 64-padded tiles, scatters bf16 output
// back to token-major order (undoing the shift).
// ---------------------------------------------------------------------------
__global__ __launch_bounds__(128) void attn_win(
    const unsigned short* __restrict__ qkv, unsigned short* __restrict__ aout) {
  __shared__ __align__(16) unsigned short sQ[64 * 72];
  __shared__ __align__(16) unsigned short sK[64 * 72];
  __shared__ __align__(16) unsigned short sV[64 * 72];  // V^T : [dim][L]
  __shared__ __align__(16) unsigned short sP[64 * 72];
  __shared__ float sS[64 * 65];

  const int wg   = blockIdx.x;
  const int head = wg % NHEAD;
  const int win  = wg / NHEAD;
  const int b    = win >> 10;         // win / 1024
  const int rem  = win & 1023;
  const int wy   = rem >> 5;
  const int wx   = rem & 31;

  const int lane = threadIdx.x & 31;
  const int wv   = threadIdx.x >> 5;  // 0..3

  // --- phase 1: build Q, K (rope+norm) and V^T tiles in LDS -----------------
  const float inv = __expf(-(float)(lane & 15) * (1.0f / 16.0f) * 4.6051702f);
  for (int r = wv; r < 64; r += 4) {
    if (r < LWIN) {
      const int ry = r / WIN, rx = r - ry * WIN;
      int ys = wy * WIN + ry + SHFT; if (ys >= HWDIM) ys -= HWDIM;
      int xs = wx * WIN + rx + SHFT; if (xs >= HWDIM) xs -= HWDIM;
      const long base =
          ((long)(b * HWDIM * HWDIM + ys * HWDIM + xs)) * QKVN + head * HDIM;
      float q1 = bf2f(qkv[base + lane]);
      float q2 = bf2f(qkv[base + 32 + lane]);
      float k1 = bf2f(qkv[base + DIMC + lane]);
      float k2 = bf2f(qkv[base + DIMC + 32 + lane]);
      float v1 = bf2f(qkv[base + 2 * DIMC + lane]);
      float v2 = bf2f(qkv[base + 2 * DIMC + 32 + lane]);
      const float pos = (lane < 16) ? (float)ry : (float)rx;
      const float ang = pos * inv;
      const float sn = __sinf(ang), cs = __cosf(ang);
      float q1r = q1 * cs - q2 * sn, q2r = q1 * sn + q2 * cs;
      float k1r = k1 * cs - k2 * sn, k2r = k1 * sn + k2 * cs;
      float qs = q1r * q1r + q2r * q2r;
      float ks = k1r * k1r + k2r * k2r;
#pragma unroll
      for (int off = 16; off > 0; off >>= 1) {
        qs += __shfl_xor(qs, off, 32);
        ks += __shfl_xor(ks, off, 32);
      }
      const float qsc = __frsqrt_rn(fmaxf(qs, 1e-24f));
      const float ksc = __frsqrt_rn(fmaxf(ks, 1e-24f));
      sQ[r * 72 + lane]      = f2bf(q1r * qsc);
      sQ[r * 72 + 32 + lane] = f2bf(q2r * qsc);
      sK[r * 72 + lane]      = f2bf(k1r * ksc);
      sK[r * 72 + 32 + lane] = f2bf(k2r * ksc);
      sV[lane * 72 + r]        = f2bf(v1);
      sV[(lane + 32) * 72 + r] = f2bf(v2);
    } else {  // zero padding rows (Q,K) / cols (V^T)
      sQ[r * 72 + lane] = 0; sQ[r * 72 + 32 + lane] = 0;
      sK[r * 72 + lane] = 0; sK[r * 72 + 32 + lane] = 0;
      sV[lane * 72 + r] = 0; sV[(lane + 32) * 72 + r] = 0;
    }
  }
  __syncthreads();

  const int l15  = lane & 15;
  const int kb8  = (lane >> 4) * 8;
  const int kb16 = (lane >> 4) * 16;
  const int arow = wv * 16 + l15;

  // --- phase 2: S = (Q K^T) / tau, masked ----------------------------------
  v8f accS[4];
#pragma unroll
  for (int i = 0; i < 4; ++i) accS[i] = (v8f)(0.0f);
#pragma unroll
  for (int ks = 0; ks < HDIM; ks += 32) {
    const v16bf aq = ld_frag_a(sQ, arow, 72, ks + kb8);
#pragma unroll
    for (int t2 = 0; t2 < 4; ++t2) {
      const v16bf bk = ld_frag_b(sK, t2 * 16 + l15, 72, ks + kb8);
      accS[t2] = wmma_bf16(aq, bk, accS[t2]);
    }
  }
#pragma unroll
  for (int t2 = 0; t2 < 4; ++t2) {
    const int colg = t2 * 16 + l15;
    const bool masked = (colg >= LWIN);
#pragma unroll
    for (int j = 0; j < 8; ++j) {
      const int rr = wv * 16 + j + ((lane >> 4) << 3);
      sS[rr * 65 + colg] = masked ? -1e30f : accS[t2][j] * 100.0f;  // 1/tau
    }
  }
  __syncthreads();

  // --- softmax rows (threads 0..63, one row each) --------------------------
  if (threadIdx.x < 64) {
    const int rr = threadIdx.x;
    float mx = -1e30f;
#pragma unroll 8
    for (int c = 0; c < 64; ++c) mx = fmaxf(mx, sS[rr * 65 + c]);
    float sum = 0.0f;
#pragma unroll 8
    for (int c = 0; c < 64; ++c) {
      const float e = __expf(sS[rr * 65 + c] - mx);
      sS[rr * 65 + c] = e;
      sum += e;
    }
    const float rs = 1.0f / sum;
#pragma unroll 8
    for (int c = 0; c < 64; ++c) sP[rr * 72 + c] = f2bf(sS[rr * 65 + c] * rs);
  }
  __syncthreads();

  // --- phase 3: O = P @ V ---------------------------------------------------
  v8f accO[4];
#pragma unroll
  for (int i = 0; i < 4; ++i) accO[i] = (v8f)(0.0f);
#pragma unroll
  for (int ks = 0; ks < 64; ks += 32) {
    const v16bf ap = ld_frag_a(sP, arow, 72, ks + kb8);
#pragma unroll
    for (int t2 = 0; t2 < 4; ++t2) {
      const v16bf bv = ld_frag_b(sV, t2 * 16 + l15, 72, ks + kb8);
      accO[t2] = wmma_bf16(ap, bv, accO[t2]);
    }
  }

  // --- scatter output (undo shift: token (y'+3)%224, (x'+3)%224) -----------
#pragma unroll
  for (int t2 = 0; t2 < 4; ++t2) {
#pragma unroll
    for (int j = 0; j < 8; ++j) {
      const int m = wv * 16 + j + ((lane >> 4) << 3);
      if (m < LWIN) {
        const int ry = m / WIN, rx = m - ry * WIN;
        int yd = wy * WIN + ry + SHFT; if (yd >= HWDIM) yd -= HWDIM;
        int xd = wx * WIN + rx + SHFT; if (xd >= HWDIM) xd -= HWDIM;
        const long tok = (long)(b * HWDIM * HWDIM + yd * HWDIM + xd);
        const int colg = head * HDIM + t2 * 16 + l15;
        aout[tok * DIMC + colg] = f2bf(accO[t2][j]);
      }
    }
  }
}

// ---------------------------------------------------------------------------
// Launch: bf16 convert -> qkv GEMM -> windowed attention -> projection GEMM
// ---------------------------------------------------------------------------
extern "C" void kernel_launch(void* const* d_in, const int* in_sizes, int n_in,
                              void* d_out, int out_size, void* d_ws,
                              size_t ws_size, hipStream_t stream) {
  const float* x      = (const float*)d_in[0];  // (2, 50176, 768) fp32
  const float* qkv_w  = (const float*)d_in[1];  // (2304, 768) fp32
  const float* proj_w = (const float*)d_in[2];  // (768, 768) fp32
  float* out = (float*)d_out;                   // (2, 50176, 768) fp32

  unsigned short* qkv  = (unsigned short*)d_ws;            // bf16, 462 MB
  unsigned short* aout = qkv  + (size_t)NTOK * QKVN;       // bf16, 154 MB
  unsigned short* xbf  = aout + (size_t)NTOK * DIMC;       // bf16, 154 MB
  unsigned short* wqkv = xbf  + (size_t)NTOK * DIMC;       // bf16, 3.5 MB
  unsigned short* wprj = wqkv + (size_t)QKVN * DIMC;       // bf16, 1.2 MB

  // 0) one-time bf16 conversion of GEMM operands
  cvt_bf16<<<((long)NTOK * DIMC / 8 + 255) / 256, 256, 0, stream>>>(
      x, xbf, (long)NTOK * DIMC);
  cvt_bf16<<<((long)QKVN * DIMC / 8 + 255) / 256, 256, 0, stream>>>(
      qkv_w, wqkv, (long)QKVN * DIMC);
  cvt_bf16<<<((long)DIMC * DIMC / 8 + 255) / 256, 256, 0, stream>>>(
      proj_w, wprj, (long)DIMC * DIMC);

  // 1) qkv = x @ qkv_w^T   (M=100352, N=2304, K=768), bf16 out
  gemm_bt<QKVN, true>
      <<<dim3(NTOK / 128, QKVN / 128), 256, 0, stream>>>(xbf, wqkv, qkv, DIMC);

  // 2) windowed attention (2048 windows x 12 heads)
  attn_win<<<NBATCH * NWSIDE * NWSIDE * NHEAD, 128, 0, stream>>>(qkv, aout);

  // 3) out = aout @ proj_w^T  (M=100352, N=768, K=768), fp32 out
  gemm_bt<DIMC, false>
      <<<dim3(NTOK / 128, DIMC / 128), 256, 0, stream>>>(aout, wprj, out, DIMC);
}
